// SelfAttention1D_79534204387962
// MI455X (gfx1250) — matrix operand order repaired and checked by reading, not compiled, not run on hardware
//
#include <hip/hip_runtime.h>

// Self-attention, B=8, S=2048, D=256, fp32 in/out. MI455X (gfx1250, wave32).
//
// Compute-bound (~41 GFLOP vs ~34 MB of mandatory HBM traffic) -> everything
// runs through v_wmma_f32_16x16x32_f16 (f16 operands, f32 accumulate).
// Pipeline:
//   K0a: x (f32) -> xh (f16), vectorized.
//   K0b: Wq/Wk/Wv (f32) -> Wt (f16, TRANSPOSED) so projection B-fragments are
//        contiguous.
//   K1 : QKV projection GEMM, one wave per 16x16 tile, pure-f16 fragment loads.
//        Qh/Kh row-major [token][d]; V written transposed Vt[b][d][s] so the
//        PV B-fragment load is contiguous along keys.
//   K2 : flash attention, one wave per 16-query tile, 64-key blocks:
//        32 score WMMAs + 32 PV WMMAs per block, online softmax in f32,
//        P transposed C-layout -> A-layout via a private 2KB LDS tile.

typedef __attribute__((ext_vector_type(16))) _Float16 v16h;
typedef __attribute__((ext_vector_type(8)))  float    v8f;

union F16Frag { v16h v; _Float16 h[16]; };

#define NTOK   16384   // B*S
#define DDIM   256
#define SLEN   2048
#define NBATCH 8
#define KBLK   64      // keys per flash block

// ---------------------------------------------------------------------------
// K0a: convert x to f16 (4 elements / thread).
// ---------------------------------------------------------------------------
__global__ __launch_bounds__(256) void convert_x_kernel(
    const float* __restrict__ x, _Float16* __restrict__ xh)
{
    const int i = (blockIdx.x * 256 + threadIdx.x) * 4;
    const float4 v = *(const float4*)(x + i);
    _Float16 h[4] = {(_Float16)v.x, (_Float16)v.y, (_Float16)v.z, (_Float16)v.w};
    *(short4*)(xh + i) = *(const short4*)h;
}

// ---------------------------------------------------------------------------
// K0b: convert + transpose the three 256x256 weights.
// Wt[z][e][d] = W_z[d][e].  grid.x = 3*65536/256 = 768 blocks.
// ---------------------------------------------------------------------------
__global__ __launch_bounds__(256) void convert_w_kernel(
    const float* __restrict__ Wq, const float* __restrict__ Wk,
    const float* __restrict__ Wv, _Float16* __restrict__ Wt)
{
    const int gid = blockIdx.x * 256 + threadIdx.x;     // 0 .. 3*65536-1
    const int z   = gid >> 16;
    const int i   = gid & 0xFFFF;
    const int e   = i >> 8;          // out row  (original col)
    const int d   = i & 255;         // out col  (original row)
    const float* W = (z == 0) ? Wq : (z == 1) ? Wk : Wv;
    Wt[(size_t)z * DDIM * DDIM + e * DDIM + d] = (_Float16)W[d * DDIM + e];
}

// ---------------------------------------------------------------------------
// K1: Y = x @ W_z. One wave computes a 16x16 tile; 8 waves / block.
// grid.x = (NTOK/16)*(DDIM/16)/8 = 2048; grid.y = 3 (q,k,v).
// ---------------------------------------------------------------------------
__global__ __launch_bounds__(256) void qkv_proj_kernel(
    const _Float16* __restrict__ xh,   // [NTOK, 256] f16
    const _Float16* __restrict__ Wt,   // [3, 256, 256] f16 transposed
    _Float16* __restrict__ Qh,         // [NTOK, 256] row-major
    _Float16* __restrict__ Kh,         // [NTOK, 256] row-major
    _Float16* __restrict__ Vt)         // [8, 256, 2048] d-major
{
    const int lane = threadIdx.x & 31;
    const int wave = threadIdx.x >> 5;
    const int w    = blockIdx.x * 8 + wave;   // 0..16383 tiles
    const int z    = blockIdx.y;              // 0:Q 1:K 2:V
    const int mt   = w >> 4;
    const int nt   = w & 15;

    const int lm = lane & 15;
    const int lh = lane >> 4;

    const _Float16* arow = xh + (size_t)(mt * 16 + lm) * DDIM;       // A row base
    const _Float16* bcol = Wt + (size_t)z * DDIM * DDIM
                              + (size_t)(nt * 16 + lm) * DDIM;       // Wt row = W col
    const int col = nt * 16 + lm;

    v8f acc = {};
    #pragma unroll
    for (int f = 0; f < 8; ++f) {             // K = 256 in steps of 32
        F16Frag a, bf;
        const _Float16* ap = arow + f * 32 + lh * 16;
        #pragma unroll
        for (int e = 0; e < 8; ++e) a.h[e]     = ap[e];
        #pragma unroll
        for (int e = 0; e < 8; ++e) a.h[8 + e] = ap[8 + e];
        const _Float16* bp = bcol + f * 32 + lh * 16;
        #pragma unroll
        for (int e = 0; e < 16; ++e) bf.h[e] = bp[e];
        acc = __builtin_amdgcn_wmma_f32_16x16x32_f16(
                  false, a.v, false, bf.v, (short)0, acc, false, false);
    }

    if (z < 2) {
        _Float16* out = (z == 0) ? Qh : Kh;
        #pragma unroll
        for (int r = 0; r < 8; ++r)
            out[(size_t)(mt * 16 + r + 8 * lh) * DDIM + col] = (_Float16)acc[r];
    } else {
        #pragma unroll
        for (int r = 0; r < 8; ++r) {
            const int tok = mt * 16 + r + 8 * lh;
            const int b   = tok >> 11;
            const int s   = tok & (SLEN - 1);
            Vt[((size_t)b * DDIM + col) * SLEN + s] = (_Float16)acc[r];
        }
    }
}

// ---------------------------------------------------------------------------
// K2: flash attention. One wave per 16-query tile; 4 waves / block.
// grid.x = NTOK/16/4 = 256 blocks of 128 threads.
// ---------------------------------------------------------------------------
__global__ __launch_bounds__(128) void attn_kernel(
    const _Float16* __restrict__ Qh,   // [NTOK, 256]
    const _Float16* __restrict__ Kh,   // [NTOK, 256]
    const _Float16* __restrict__ Vt,   // [8, 256, 2048]
    float* __restrict__ out)           // [NTOK, 256] fp32
{
    __shared__ _Float16 plds[4][16 * KBLK];   // 8 KB: per-wave P staging

    const int lane  = threadIdx.x & 31;
    const int wave  = threadIdx.x >> 5;
    const int qt    = blockIdx.x * 4 + wave;   // 0..1023
    const int b     = qt >> 7;
    const int qbase = (qt & 127) * 16;
    const int tokq  = b * SLEN + qbase;

    const int lm = lane & 15;
    const int lh = lane >> 4;

    v8f O[16];
    #pragma unroll
    for (int n = 0; n < 16; ++n) { v8f z = {}; O[n] = z; }
    float mrow[8], lrow[8];
    #pragma unroll
    for (int r = 0; r < 8; ++r) { mrow[r] = -__builtin_huge_valf(); lrow[r] = 0.0f; }

    _Float16* P = plds[wave];
    const _Float16* qrow   = Qh + (size_t)(tokq + lm) * DDIM;
    const _Float16* kbatch = Kh + (size_t)b * SLEN * DDIM;
    const _Float16* vbatch = Vt + (size_t)b * DDIM * SLEN;

    for (int kb = 0; kb < SLEN / KBLK; ++kb) {
        const int keybase = kb * KBLK;

        // Prefetch next key block's K rows (2 lines per lane covers 64 rows).
        if (kb + 1 < SLEN / KBLK) {
            const _Float16* nk = kbatch + (size_t)(keybase + KBLK + lane) * DDIM;
            __builtin_prefetch(nk, 0, 3);
            __builtin_prefetch(nk + 32 * DDIM, 0, 3);
        }

        // ---- scores: four 16x16 tiles covering 64 keys ----------------------
        v8f s[4];
        #pragma unroll
        for (int t = 0; t < 4; ++t) {
            v8f acc = {};
            const _Float16* krow = kbatch + (size_t)(keybase + t * 16 + lm) * DDIM;
            #pragma unroll
            for (int f = 0; f < 8; ++f) {
                F16Frag a, bf;
                const _Float16* ap = qrow + f * 32 + lh * 16;
                #pragma unroll
                for (int e = 0; e < 8; ++e) a.h[e]     = ap[e];
                #pragma unroll
                for (int e = 0; e < 8; ++e) a.h[8 + e] = ap[8 + e];
                const _Float16* bp = krow + f * 32 + lh * 16;
                #pragma unroll
                for (int e = 0; e < 16; ++e) bf.h[e] = bp[e];
                acc = __builtin_amdgcn_wmma_f32_16x16x32_f16(
                          false, a.v, false, bf.v, (short)0, acc, false, false);
            }
            #pragma unroll
            for (int r = 0; r < 8; ++r) acc[r] *= 0.0625f;   // 1/sqrt(256)
            s[t] = acc;
        }

        // ---- online softmax (rows replicated across each half-wave) ---------
        #pragma unroll
        for (int r = 0; r < 8; ++r) {
            float v = fmaxf(fmaxf(s[0][r], s[1][r]), fmaxf(s[2][r], s[3][r]));
            v = fmaxf(v, __shfl_xor(v, 1, 32));
            v = fmaxf(v, __shfl_xor(v, 2, 32));
            v = fmaxf(v, __shfl_xor(v, 4, 32));
            v = fmaxf(v, __shfl_xor(v, 8, 32));
            const float mnew = fmaxf(mrow[r], v);
            const float corr = __expf(mrow[r] - mnew);   // 0 on first block
            mrow[r] = mnew;

            float rsum = 0.0f;
            #pragma unroll
            for (int t = 0; t < 4; ++t) {
                const float p = __expf(s[t][r] - mnew);
                s[t][r] = p;
                rsum += p;
            }
            rsum += __shfl_xor(rsum, 1, 32);
            rsum += __shfl_xor(rsum, 2, 32);
            rsum += __shfl_xor(rsum, 4, 32);
            rsum += __shfl_xor(rsum, 8, 32);
            lrow[r] = lrow[r] * corr + rsum;

            #pragma unroll
            for (int n = 0; n < 16; ++n) O[n][r] *= corr;
        }

        // ---- P: C-layout -> LDS (row-major 16x64 f16) -> A-fragments --------
        #pragma unroll
        for (int t = 0; t < 4; ++t)
            #pragma unroll
            for (int r = 0; r < 8; ++r)
                P[(r + 8 * lh) * KBLK + t * 16 + lm] = (_Float16)s[t][r];
        // same-wave DS ops are in-order; compiler inserts s_wait_dscnt for RAW
        F16Frag pa[2];
        #pragma unroll
        for (int g = 0; g < 2; ++g) {
            const _Float16* pp = P + lm * KBLK + g * 32 + lh * 16;
            #pragma unroll
            for (int e = 0; e < 8; ++e) pa[g].h[e]     = pp[e];
            #pragma unroll
            for (int e = 0; e < 8; ++e) pa[g].h[8 + e] = pp[8 + e];
        }

        // ---- O += P @ V : 16 d-tiles x 2 K=32 WMMAs -------------------------
        #pragma unroll
        for (int n = 0; n < 16; ++n) {
            const _Float16* vp0 = vbatch + (size_t)(n * 16 + lm) * SLEN + keybase;
            #pragma unroll
            for (int g = 0; g < 2; ++g) {
                F16Frag vf;
                const _Float16* vp = vp0 + g * 32 + lh * 16;
                #pragma unroll
                for (int e = 0; e < 16; ++e) vf.h[e] = vp[e];
                O[n] = __builtin_amdgcn_wmma_f32_16x16x32_f16(
                           false, pa[g].v, false, vf.v, (short)0, O[n], false, false);
            }
        }
    }

    // ---- epilogue: normalize, store fp32 ------------------------------------
    #pragma unroll
    for (int r = 0; r < 8; ++r) {
        const float inv = 1.0f / lrow[r];
        float* orow = out + (size_t)(tokq + r + 8 * lh) * DDIM;
        #pragma unroll
        for (int n = 0; n < 16; ++n)
            orow[n * 16 + lm] = O[n][r] * inv;
    }
}

// ---------------------------------------------------------------------------
extern "C" void kernel_launch(void* const* d_in, const int* in_sizes, int n_in,
                              void* d_out, int out_size, void* d_ws, size_t ws_size,
                              hipStream_t stream) {
    const float* x  = (const float*)d_in[0];
    const float* Wq = (const float*)d_in[1];
    const float* Wk = (const float*)d_in[2];
    const float* Wv = (const float*)d_in[3];

    _Float16* xh = (_Float16*)d_ws;                    //  8 MiB
    _Float16* Qh = xh + (size_t)NTOK * DDIM;           //  8 MiB
    _Float16* Kh = Qh + (size_t)NTOK * DDIM;           //  8 MiB
    _Float16* Vt = Kh + (size_t)NTOK * DDIM;           //  8 MiB
    _Float16* Wt = Vt + (size_t)NTOK * DDIM;           //  384 KiB (total ~32.4 MiB)

    convert_x_kernel<<<(NTOK * DDIM) / 4 / 256, 256, 0, stream>>>(x, xh);
    convert_w_kernel<<<3 * DDIM * DDIM / 256, 256, 0, stream>>>(Wq, Wk, Wv, Wt);

    dim3 g1(2048, 3);
    qkv_proj_kernel<<<g1, 256, 0, stream>>>(xh, Wt, Qh, Kh, Vt);

    attn_kernel<<<256, 128, 0, stream>>>(Qh, Kh, Vt, (float*)d_out);
}
